// ImprovedGNNModel_66614942761339
// MI455X (gfx1250) — matrix operand order repaired and checked by reading, not compiled, not run on hardware
//
#include <hip/hip_runtime.h>

// ---------------------------------------------------------------------------
// MI455X (gfx1250) implementation of the GNN reference.
// All dense GEMMs run on v_wmma_f32_16x16x32_f16 (wave32, f32 accumulate).
// Edge softmax / scatter use float atomics (L2-resident gathers).
// ---------------------------------------------------------------------------

#define NNODES 20000
#define NEDGES 320000
#define DCH    128
#define NHEAD  4
#define NGRAPH 64

typedef __attribute__((ext_vector_type(16))) _Float16 v16h;
typedef __attribute__((ext_vector_type(8)))  _Float16 v8h;
typedef __attribute__((ext_vector_type(8)))  float    v8f;

// ---------------------------------------------------------------------------
// WMMA GEMM: C[M,N] = act( A[M,K](f16,row-major) * Wp(packed f16) + bias )
// block = 64 threads (2 waves); each wave computes a 16(M) x 64(N) tile.
// grid = (M/16, N/128).  M,N multiples of 16/128 -> no divergence, EXEC=~0.
// Packed B layout per 32(K)x16(N) tile: elem(lane, j) = W[kt*32 + (lane>=16)*8 + (j&7) + (j>=8)*16][nt*16 + (lane&15)]
// ---------------------------------------------------------------------------
__global__ void k_wmma_gemm(const _Float16* __restrict__ A,
                            const _Float16* __restrict__ Wp,
                            const float* __restrict__ bias,
                            float* __restrict__ C,
                            int M, int Nn, int Kk, int act) {
  const int lane = threadIdx.x & 31;
  const int wave = threadIdx.x >> 5;
  const int mt = blockIdx.x;
  const int colbase = blockIdx.y * 128 + wave * 64;
  const int NT = Nn >> 4;
  const int KT = Kk >> 5;
  const int ntb = colbase >> 4;

  const int mrow = mt * 16 + (lane & 15);
  const int ksub = (lane >> 4) << 3;      // 0 or 8 (A-fragment K sub-offset)
  const _Float16* arow = A + (size_t)mrow * Kk;

  v8f acc[4] = {v8f{}, v8f{}, v8f{}, v8f{}};

  for (int kt = 0; kt < KT; ++kt) {
    const int kbase = (kt << 5) + ksub;
    v8h a0 = *(const v8h*)(arow + kbase);
    v8h a1 = *(const v8h*)(arow + kbase + 16);
    v16h af;
#pragma unroll
    for (int i = 0; i < 8; ++i) { af[i] = a0[i]; af[i + 8] = a1[i]; }
#pragma unroll
    for (int t = 0; t < 4; ++t) {
      const _Float16* bp = Wp + (((size_t)kt * NT + (ntb + t)) * 32 + lane) * 16;
      v8h b0 = *(const v8h*)(bp);
      v8h b1 = *(const v8h*)(bp + 8);
      v16h bf;
#pragma unroll
      for (int i = 0; i < 8; ++i) { bf[i] = b0[i]; bf[i + 8] = b1[i]; }
      acc[t] = __builtin_amdgcn_wmma_f32_16x16x32_f16(false, af, false, bf,
                                                      (short)0, acc[t],
                                                      false, false);
    }
  }

  const int rowoff = (lane >> 4) << 3;
#pragma unroll
  for (int t = 0; t < 4; ++t) {
    const int n = colbase + t * 16 + (lane & 15);
    const float bs = bias ? bias[n] : 0.0f;
#pragma unroll
    for (int r = 0; r < 8; ++r) {
      float v = acc[t][r] + bs;
      if (act == 1) {          // SELU
        v = 1.0507009873554805f *
            (v > 0.0f ? v : 1.6732632423543772f * (__expf(v) - 1.0f));
      } else if (act == 2) {   // sigmoid
        v = 1.0f / (1.0f + __expf(-v));
      }
      C[(size_t)(mt * 16 + rowoff + r) * Nn + n] = v;
    }
  }
}

// Pack f32 row-major W[K,N] into fragment-ready f16 tiles.
__global__ void k_pack_weight(const float* __restrict__ W, _Float16* __restrict__ Wp,
                              int Kk, int Nn) {
  int idx = blockIdx.x * blockDim.x + threadIdx.x;
  int total = (Kk >> 5) * (Nn >> 4) * 512;
  if (idx >= total) return;
  int j    = idx & 15;
  int lane = (idx >> 4) & 31;
  int tile = idx >> 9;
  int NT = Nn >> 4;
  int nt = tile % NT;
  int kt = tile / NT;
  int n = nt * 16 + (lane & 15);
  int k = kt * 32 + ((lane >> 4) << 3) + (j & 7) + ((j >> 3) << 4);
  Wp[idx] = (_Float16)W[(size_t)k * Nn + n];
}

// ---------------------------------------------------------------------------
// elementwise helpers
// ---------------------------------------------------------------------------
__global__ void k_fill(float* p, float v, int n) {
  int i = blockIdx.x * blockDim.x + threadIdx.x;
  if (i < n) p[i] = v;
}
__global__ void k_f32_to_f16(const float* __restrict__ a, _Float16* __restrict__ o, int n) {
  int i = blockIdx.x * blockDim.x + threadIdx.x;
  if (i < n) o[i] = (_Float16)a[i];
}
__global__ void k_add_f16(const float* __restrict__ a, const float* __restrict__ b,
                          _Float16* __restrict__ o, int n) {
  int i = blockIdx.x * blockDim.x + threadIdx.x;
  if (i < n) o[i] = (_Float16)(a[i] + b[i]);
}
__global__ void k_mul_f16(const float* __restrict__ a, const float* __restrict__ b,
                          _Float16* __restrict__ o, int n) {
  int i = blockIdx.x * blockDim.x + threadIdx.x;
  if (i < n) o[i] = (_Float16)(a[i] * b[i]);
}

// ---------------------------------------------------------------------------
// edge attention
// ---------------------------------------------------------------------------
__device__ inline void atomicMaxF(float* addr, float val) {
  if (val >= 0.0f) atomicMax((int*)addr, __float_as_int(val));
  else             atomicMin((unsigned int*)addr, __float_as_uint(val));
}

// one wave32 per edge; 4 heads of 128-dim dot products
__global__ void k_edge_logits(const int* __restrict__ ei, const float* __restrict__ ea,
                              const float* __restrict__ Q, const float* __restrict__ Kv,
                              const float* __restrict__ we,
                              float* __restrict__ logits, float* __restrict__ mmax) {
  int wid = (blockIdx.x * blockDim.x + threadIdx.x) >> 5;
  int lane = threadIdx.x & 31;
  if (wid >= NEDGES) return;
  int src = ei[wid];
  int dst = ei[NEDGES + wid];
  float a = ea[wid];
  const float4* qr = (const float4*)(Q + (size_t)dst * 512);
  const float4* kr = (const float4*)(Kv + (size_t)src * 512);
  const float4* wr = (const float4*)we;
#pragma unroll
  for (int h = 0; h < NHEAD; ++h) {
    float4 qv = qr[h * 32 + lane];
    float4 kv = kr[h * 32 + lane];
    float4 wv = wr[h * 32 + lane];
    float s = qv.x * (kv.x + a * wv.x) + qv.y * (kv.y + a * wv.y) +
              qv.z * (kv.z + a * wv.z) + qv.w * (kv.w + a * wv.w);
#pragma unroll
    for (int o = 16; o > 0; o >>= 1) s += __shfl_xor(s, o, 32);
    if (lane == 0) {
      float lg = s * 0.088388347648318447f;  // 1/sqrt(128)
      logits[(size_t)wid * NHEAD + h] = lg;
      atomicMaxF(&mmax[dst * NHEAD + h], lg);
    }
  }
}

__global__ void k_edge_exp(const int* __restrict__ ei, float* __restrict__ logits,
                           const float* __restrict__ mmax, float* __restrict__ den) {
  int idx = blockIdx.x * blockDim.x + threadIdx.x;
  if (idx >= NEDGES * NHEAD) return;
  int e = idx >> 2;
  int h = idx & 3;
  int dst = ei[NEDGES + e];
  float ex = __expf(logits[idx] - mmax[dst * NHEAD + h]);
  logits[idx] = ex;  // in place: logits buffer becomes exp buffer
  atomicAdd(&den[dst * NHEAD + h], ex);
}

// one wave32 per edge: out[dst] += alpha * (v[src] + ea*we)
__global__ void k_edge_scatter(const int* __restrict__ ei, const float* __restrict__ ea,
                               const float* __restrict__ V, const float* __restrict__ we,
                               const float* __restrict__ exv, const float* __restrict__ den,
                               float* __restrict__ out) {
  int wid = (blockIdx.x * blockDim.x + threadIdx.x) >> 5;
  int lane = threadIdx.x & 31;
  if (wid >= NEDGES) return;
  int src = ei[wid];
  int dst = ei[NEDGES + wid];
  float a = ea[wid];
  const float4* vr = (const float4*)(V + (size_t)src * 512);
  const float4* wr = (const float4*)we;
#pragma unroll
  for (int h = 0; h < NHEAD; ++h) {
    float alpha = exv[(size_t)wid * NHEAD + h] / den[dst * NHEAD + h];
    float4 vv = vr[h * 32 + lane];
    float4 wv = wr[h * 32 + lane];
    float* op = out + (size_t)dst * 512 + h * 128 + lane * 4;
    atomicAdd(op + 0, alpha * (vv.x + a * wv.x));
    atomicAdd(op + 1, alpha * (vv.y + a * wv.y));
    atomicAdd(op + 2, alpha * (vv.z + a * wv.z));
    atomicAdd(op + 3, alpha * (vv.w + a * wv.w));
  }
}

__global__ void k_headmean_skip(const float* __restrict__ att, const float* __restrict__ skip,
                                float* __restrict__ outc) {
  int idx = blockIdx.x * blockDim.x + threadIdx.x;
  if (idx >= NNODES * DCH) return;
  int n = idx >> 7;
  int c = idx & 127;
  const float* r = att + (size_t)n * 512;
  outc[idx] = 0.25f * (r[c] + r[128 + c] + r[256 + c] + r[384 + c]) + skip[idx];
}

// ---------------------------------------------------------------------------
// pooling + gating
// ---------------------------------------------------------------------------
__global__ void k_pool(const float* __restrict__ x, const int* __restrict__ batch,
                       float* __restrict__ sums, float* __restrict__ cnt) {
  int idx = blockIdx.x * blockDim.x + threadIdx.x;
  if (idx >= NNODES * DCH) return;
  int n = idx >> 7;
  int c = idx & 127;
  int g = batch[n];
  atomicAdd(&sums[g * DCH + c], x[idx]);
  if (c == 0) atomicAdd(&cnt[g], 1.0f);
}

__global__ void k_pool_div_f16(const float* __restrict__ sums, const float* __restrict__ cnt,
                               _Float16* __restrict__ gm16) {
  int idx = blockIdx.x * blockDim.x + threadIdx.x;
  if (idx >= NGRAPH * DCH) return;
  int g = idx >> 7;
  gm16[idx] = (_Float16)(sums[idx] / fmaxf(cnt[g], 1.0f));
}

__global__ void k_gate_in(const float* __restrict__ tx, const float* __restrict__ mg,
                          const int* __restrict__ batch, const float* __restrict__ alpha,
                          _Float16* __restrict__ gi) {
  int idx = blockIdx.x * blockDim.x + threadIdx.x;
  if (idx >= NNODES * DCH) return;
  int n = idx >> 7;
  int c = idx & 127;
  gi[(size_t)n * 256 + c]       = (_Float16)tx[idx];
  gi[(size_t)n * 256 + 128 + c] = (_Float16)(alpha[0] * mg[batch[n] * DCH + c]);
}

__global__ void k_combine(const float* __restrict__ tx, const float* __restrict__ mg,
                          const int* __restrict__ batch, const float* __restrict__ gate,
                          const float* __restrict__ belta, float* __restrict__ outc) {
  int idx = blockIdx.x * blockDim.x + threadIdx.x;
  if (idx >= NNODES * DCH) return;
  int n = idx >> 7;
  int c = idx & 127;
  float g = gate[idx];
  float m = mg[batch[n] * DCH + c];
  float t = tx[idx];
  outc[idx] = t + belta[0] * (g * t + (1.0f - g) * m);
}

// ---------------------------------------------------------------------------
// host orchestration
// ---------------------------------------------------------------------------
static inline int cdiv(int a, int b) { return (a + b - 1) / b; }

static void gemm(const _Float16* A, const _Float16* Wp, const float* bias, float* C,
                 int M, int Nn, int Kk, int act, hipStream_t s) {
  dim3 grid(M / 16, Nn / 128);
  k_wmma_gemm<<<grid, 64, 0, s>>>(A, Wp, bias, C, M, Nn, Kk, act);
}
static void pack(const float* W, _Float16* Wp, int Kk, int Nn, hipStream_t s) {
  int total = (Kk / 32) * (Nn / 16) * 512;
  k_pack_weight<<<cdiv(total, 256), 256, 0, s>>>(W, Wp, Kk, Nn);
}

struct ConvW {
  const float *wq, *bq, *wk, *bk, *wv, *bv, *we, *wskip, *bskip;
  _Float16 *wq_p, *wk_p, *wv_p, *wskip_p;
};
struct SgW {
  const float *w1, *b1, *w2, *b2, *wo, *bo;
  _Float16 *w1_p, *w2_p, *wo_p;
};

extern "C" void kernel_launch(void* const* d_in, const int* in_sizes, int n_in,
                              void* d_out, int out_size, void* d_ws, size_t ws_size,
                              hipStream_t stream) {
  (void)in_sizes; (void)n_in; (void)out_size; (void)ws_size;

  // ---- inputs (setup_inputs() dict order, params depth-first in insertion order)
  const float* tx    = (const float*)d_in[0];   // [N,128]
  const int*   ei    = (const int*)  d_in[1];   // [2,E]
  const float* ea    = (const float*)d_in[2];   // [E]
  const int*   batch = (const int*)  d_in[3];   // [N]

  ConvW c1 = {(const float*)d_in[4],  (const float*)d_in[5],  (const float*)d_in[6],
              (const float*)d_in[7],  (const float*)d_in[8],  (const float*)d_in[9],
              (const float*)d_in[10], (const float*)d_in[11], (const float*)d_in[12],
              nullptr, nullptr, nullptr, nullptr};
  ConvW c2 = {(const float*)d_in[13], (const float*)d_in[14], (const float*)d_in[15],
              (const float*)d_in[16], (const float*)d_in[17], (const float*)d_in[18],
              (const float*)d_in[19], (const float*)d_in[20], (const float*)d_in[21],
              nullptr, nullptr, nullptr, nullptr};
  SgW sg1 = {(const float*)d_in[22], (const float*)d_in[23], (const float*)d_in[24],
             (const float*)d_in[25], (const float*)d_in[26], (const float*)d_in[27],
             nullptr, nullptr, nullptr};
  SgW sg2 = {(const float*)d_in[28], (const float*)d_in[29], (const float*)d_in[30],
             (const float*)d_in[31], (const float*)d_in[32], (const float*)d_in[33],
             nullptr, nullptr, nullptr};
  const float* wmap  = (const float*)d_in[34];
  const float* bmap  = (const float*)d_in[35];
  const float* wgate = (const float*)d_in[36];
  const float* bgate = (const float*)d_in[37];
  const float* alpha = (const float*)d_in[38];
  const float* belta = (const float*)d_in[39];

  // ---- workspace bump allocator
  char* ws = (char*)d_ws;
  size_t off = 0;
  auto alloc = [&](size_t bytes) -> char* {
    char* p = ws + off;
    off = (off + bytes + 255) & ~(size_t)255;
    return p;
  };
  _Float16* TX16   = (_Float16*)alloc((size_t)NNODES * DCH * 2);
  _Float16* XCUR16 = (_Float16*)alloc((size_t)NNODES * DCH * 2);
  _Float16* RES16  = (_Float16*)alloc((size_t)NNODES * DCH * 2);
  float*    Qb     = (float*)alloc((size_t)NNODES * 512 * 4);   // also swiglu 'a'
  float*    Kb     = (float*)alloc((size_t)NNODES * 512 * 4);   // also swiglu selu(b)
  float*    Vb     = (float*)alloc((size_t)NNODES * 512 * 4);
  float*    ATT    = (float*)alloc((size_t)NNODES * 512 * 4);
  float*    SKIP   = (float*)alloc((size_t)NNODES * DCH * 4);
  float*    OUTC   = (float*)alloc((size_t)NNODES * DCH * 4);
  float*    XCUR   = (float*)alloc((size_t)NNODES * DCH * 4);
  float*    GATE   = (float*)alloc((size_t)NNODES * DCH * 4);
  float*    LOGITS = (float*)alloc((size_t)NEDGES * NHEAD * 4); // then exp in-place
  float*    MMAX   = (float*)alloc((size_t)NNODES * NHEAD * 4);
  float*    DEN    = (float*)alloc((size_t)NNODES * NHEAD * 4);
  _Float16* H16    = (_Float16*)alloc((size_t)NNODES * 512 * 2); // hidden / gate_in
  float*    SUMS   = (float*)alloc((size_t)NGRAPH * DCH * 4);
  float*    CNT    = (float*)alloc((size_t)NGRAPH * 4);
  _Float16* GM16   = (_Float16*)alloc((size_t)NGRAPH * DCH * 2);
  // packed f16 weights
  _Float16* WPBASE = (_Float16*)alloc((size_t)900000 * 2);
  size_t wpo = 0;
  auto wpAlloc = [&](size_t elems) -> _Float16* { _Float16* p = WPBASE + wpo; wpo += elems; return p; };
  c1.wq_p = wpAlloc(128 * 512); c1.wk_p = wpAlloc(128 * 512);
  c1.wv_p = wpAlloc(128 * 512); c1.wskip_p = wpAlloc(128 * 128);
  c2.wq_p = wpAlloc(128 * 512); c2.wk_p = wpAlloc(128 * 512);
  c2.wv_p = wpAlloc(128 * 512); c2.wskip_p = wpAlloc(128 * 128);
  sg1.w1_p = wpAlloc(128 * 512); sg1.w2_p = wpAlloc(128 * 512); sg1.wo_p = wpAlloc(512 * 128);
  sg2.w1_p = wpAlloc(128 * 512); sg2.w2_p = wpAlloc(128 * 512); sg2.wo_p = wpAlloc(512 * 128);
  _Float16* wmap_p  = wpAlloc(128 * 128);
  _Float16* wgate_p = wpAlloc(256 * 128);

  // ---- pack all weights to fragment-ready f16
  pack(c1.wq, c1.wq_p, 128, 512, stream);  pack(c1.wk, c1.wk_p, 128, 512, stream);
  pack(c1.wv, c1.wv_p, 128, 512, stream);  pack(c1.wskip, c1.wskip_p, 128, 128, stream);
  pack(c2.wq, c2.wq_p, 128, 512, stream);  pack(c2.wk, c2.wk_p, 128, 512, stream);
  pack(c2.wv, c2.wv_p, 128, 512, stream);  pack(c2.wskip, c2.wskip_p, 128, 128, stream);
  pack(sg1.w1, sg1.w1_p, 128, 512, stream); pack(sg1.w2, sg1.w2_p, 128, 512, stream);
  pack(sg1.wo, sg1.wo_p, 512, 128, stream);
  pack(sg2.w1, sg2.w1_p, 128, 512, stream); pack(sg2.w2, sg2.w2_p, 128, 512, stream);
  pack(sg2.wo, sg2.wo_p, 512, 128, stream);
  pack(wmap, wmap_p, 128, 128, stream);
  pack(wgate, wgate_p, 256, 128, stream);

  const int NE = NNODES * DCH;            // 2.56M
  k_f32_to_f16<<<cdiv(NE, 256), 256, 0, stream>>>(tx, TX16, NE);

  auto run_conv = [&](const ConvW& p, const _Float16* Xin16) {
    gemm(Xin16, p.wq_p, p.bq, Qb, NNODES, 512, 128, 0, stream);
    gemm(Xin16, p.wk_p, p.bk, Kb, NNODES, 512, 128, 0, stream);
    gemm(Xin16, p.wv_p, p.bv, Vb, NNODES, 512, 128, 0, stream);
    gemm(Xin16, p.wskip_p, p.bskip, SKIP, NNODES, 128, 128, 0, stream);
    k_fill<<<cdiv(NNODES * NHEAD, 256), 256, 0, stream>>>(MMAX, -3.0e38f, NNODES * NHEAD);
    k_fill<<<cdiv(NNODES * NHEAD, 256), 256, 0, stream>>>(DEN, 0.0f, NNODES * NHEAD);
    k_fill<<<cdiv(NNODES * 512, 256), 256, 0, stream>>>(ATT, 0.0f, NNODES * 512);
    k_edge_logits<<<NEDGES / 8, 256, 0, stream>>>(ei, ea, Qb, Kb, p.we, LOGITS, MMAX);
    k_edge_exp<<<cdiv(NEDGES * NHEAD, 256), 256, 0, stream>>>(ei, LOGITS, MMAX, DEN);
    k_edge_scatter<<<NEDGES / 8, 256, 0, stream>>>(ei, ea, Vb, p.we, LOGITS, DEN, ATT);
    k_headmean_skip<<<cdiv(NE, 256), 256, 0, stream>>>(ATT, SKIP, OUTC);
  };
  auto run_swiglu = [&](const SgW& p, const _Float16* Xin16, float* outF32) {
    gemm(Xin16, p.w1_p, p.b1, Qb, NNODES, 512, 128, 0, stream);      // a
    gemm(Xin16, p.w2_p, p.b2, Kb, NNODES, 512, 128, 1, stream);      // selu(b)
    k_mul_f16<<<cdiv(NNODES * 512, 256), 256, 0, stream>>>(Qb, Kb, H16, NNODES * 512);
    gemm(H16, p.wo_p, p.bo, outF32, NNODES, 128, 512, 0, stream);
  };

  // layer 1
  run_conv(c1, TX16);
  k_add_f16<<<cdiv(NE, 256), 256, 0, stream>>>(tx, OUTC, RES16, NE);
  run_swiglu(sg1, RES16, XCUR);
  k_f32_to_f16<<<cdiv(NE, 256), 256, 0, stream>>>(XCUR, XCUR16, NE);
  // layer 2
  run_conv(c2, XCUR16);
  k_add_f16<<<cdiv(NE, 256), 256, 0, stream>>>(tx, OUTC, RES16, NE);
  run_swiglu(sg2, RES16, XCUR);   // XCUR now holds final node features

  // global mean pool + map
  k_fill<<<cdiv(NGRAPH * DCH, 256), 256, 0, stream>>>(SUMS, 0.0f, NGRAPH * DCH);
  k_fill<<<1, 64, 0, stream>>>(CNT, 0.0f, NGRAPH);
  k_pool<<<cdiv(NE, 256), 256, 0, stream>>>(XCUR, batch, SUMS, CNT);
  k_pool_div_f16<<<cdiv(NGRAPH * DCH, 256), 256, 0, stream>>>(SUMS, CNT, GM16);

  float* outG = (float*)d_out;                 // mapped_global [G,128]
  float* outC = outG + NGRAPH * DCH;           // combined [N,128]
  gemm(GM16, wmap_p, bmap, outG, NGRAPH, 128, 128, 0, stream);

  // gating
  k_gate_in<<<cdiv(NE, 256), 256, 0, stream>>>(tx, outG, batch, alpha, H16);
  gemm(H16, wgate_p, bgate, GATE, NNODES, 128, 256, 2, stream);
  k_combine<<<cdiv(NE, 256), 256, 0, stream>>>(tx, outG, batch, GATE, belta, outC);
}
